// GroupQueryAttention_39436389712547
// MI455X (gfx1250) — matrix operand order repaired and checked by reading, not compiled, not run on hardware
//
#include <hip/hip_runtime.h>
#include <hip/hip_bf16.h>

typedef __attribute__((ext_vector_type(16))) _Float16 v16h;
typedef __attribute__((ext_vector_type(8)))  float    v8f;

#define S_LEN 2048
#define HDIM  1024
#define NQ    16
#define NKV   4
#define DHEAD 64
#define KVDIM (NKV * DHEAD)   // 256
#define SCALE 0.125f          // 1/sqrt(64)

// ---------------------------------------------------------------------------
// WMMA operand loaders (LDS row-major -> WMMA lane layouts, contiguous k, 16-bit data).
// A 16x32 (MxK): lanes 0-15 row M=lane hold K 0..15; lanes 16-31 row
// M=lane-16 hold K 16..31 (same k order as the B loaders).
// ---------------------------------------------------------------------------
__device__ __forceinline__ v16h wmma_load_a(const _Float16* lds, int stride,
                                            int m0, int k0) {
  const int lane = threadIdx.x & 31;
  const int hi   = lane >> 4;          // which half-wave
  const int r    = lane & 15;
  const _Float16* p = lds + (m0 + r) * stride + k0 + hi * 16;
  v16h a;
#pragma unroll
  for (int i = 0; i < 8; ++i) a[i] = p[i];
#pragma unroll
  for (int i = 0; i < 8; ++i) a[8 + i] = p[8 + i];
  return a;
}

// B 32x16 (KxN) built from row-major [N][K] storage (i.e. B = rowsᵀ):
// lanes 0-15 col N=lane hold K 0..15; lanes 16-31 col N=lane-16 hold K 16..31.
__device__ __forceinline__ v16h wmma_load_bt(const _Float16* lds, int stride,
                                             int n0, int k0) {
  const int lane = threadIdx.x & 31;
  const int hi   = lane >> 4;
  const int n    = lane & 15;
  const _Float16* p = lds + (n0 + n) * stride + k0 + hi * 16;
  v16h b;
#pragma unroll
  for (int i = 0; i < 16; ++i) b[i] = p[i];
  return b;
}

// B 32x16 (KxN) from row-major [K][N] storage (direct, used for P·V).
__device__ __forceinline__ v16h wmma_load_bn(const _Float16* lds, int stride,
                                             int k0, int n0) {
  const int lane = threadIdx.x & 31;
  const int hi   = lane >> 4;
  const int n    = lane & 15;
  const _Float16* p = lds + (k0 + hi * 16) * stride + n0 + n;
  v16h b;
#pragma unroll
  for (int i = 0; i < 16; ++i) b[i] = p[i * stride];
  return b;
}

__device__ __forceinline__ v8f wmma_f16(v16h a, v16h b, v8f c) {
  return __builtin_amdgcn_wmma_f32_16x16x32_f16(false, a, false, b,
                                                (short)0, c, false, false);
}

// ---------------------------------------------------------------------------
// Async tile copy, CDNA5 path: GLOBAL_LOAD_ASYNC_TO_LDS_B128 (ASYNCcnt).
// 64 rows x 64 halves (128 B/row); 128 threads issue 4 x 16 B chunks each.
// Generic LDS pointer: addr[31:0] is the LDS byte offset (ISA aperture rule),
// which is exactly the VDST operand of the async op.
// ---------------------------------------------------------------------------
__device__ __forceinline__ void async_tile64(const _Float16* g, int gstride,
                                             _Float16* lds) {
  const int t = threadIdx.x;            // 0..127
#pragma unroll
  for (int i = 0; i < 4; ++i) {
    const int c   = t + i * 128;        // chunk id 0..511
    const int row = c >> 3;             // 8 chunks per row
    const int col = (c & 7) * 8;        // halves
    unsigned int ldsaddr = (unsigned int)(uintptr_t)(lds + row * 64 + col);
    const _Float16* gaddr = g + (size_t)row * gstride + col;
    asm volatile("global_load_async_to_lds_b128 %0, %1, off"
                 :: "v"(ldsaddr), "v"(gaddr)
                 : "memory");
  }
}

__device__ __forceinline__ void wait_async_all() {
  asm volatile("s_wait_asynccnt 0x0" ::: "memory");
}

// ---------------------------------------------------------------------------
// f32 -> f16 conversion
// ---------------------------------------------------------------------------
__global__ void cvt_f32_f16(const float* __restrict__ src,
                            _Float16* __restrict__ dst, int n) {
  int i = blockIdx.x * blockDim.x + threadIdx.x;
  if (i < n) dst[i] = (_Float16)src[i];
}

// ---------------------------------------------------------------------------
// GEMM  C[M x N] = A[M x K] * W[N x K]^T   (f16 in, f32 out)
// Block: 128 threads = 4 waves; each block makes a 64x64 C tile,
// each wave owns 16 rows x 64 cols.
// ---------------------------------------------------------------------------
__global__ __launch_bounds__(128)
void gemm_nt_f16(const _Float16* __restrict__ A, const _Float16* __restrict__ W,
                 float* __restrict__ C, int M, int N, int K) {
  __shared__ _Float16 As[64 * 64];
  __shared__ _Float16 Ws[64 * 64];

  const int mbase = blockIdx.y * 64;
  const int nbase = blockIdx.x * 64;
  const int wave  = threadIdx.x >> 5;
  const int m0    = wave * 16;

  v8f acc[4];
#pragma unroll
  for (int t = 0; t < 4; ++t)
#pragma unroll
    for (int e = 0; e < 8; ++e) acc[t][e] = 0.0f;

  for (int kb = 0; kb < K; kb += 64) {
    __syncthreads();                          // prior reads done
    async_tile64(A + (size_t)mbase * K + kb, K, As);
    async_tile64(W + (size_t)nbase * K + kb, K, Ws);
    wait_async_all();                         // own async writes landed
    __syncthreads();                          // everyone's writes visible
#pragma unroll
    for (int kc = 0; kc < 2; ++kc) {
      v16h a = wmma_load_a(As, 64, m0, kc * 32);
#pragma unroll
      for (int t = 0; t < 4; ++t) {
        v16h b = wmma_load_bt(Ws, 64, t * 16, kc * 32);
        acc[t] = wmma_f16(a, b, acc[t]);
      }
    }
  }

  // C layout: VGPR r -> row m0+r (lanes 0-15) / m0+r+8 (lanes 16-31)
  const int lane = threadIdx.x & 31;
  const int hi   = lane >> 4;
  const int nl   = lane & 15;
#pragma unroll
  for (int t = 0; t < 4; ++t)
#pragma unroll
    for (int r = 0; r < 8; ++r) {
      int row = mbase + m0 + r + hi * 8;
      int col = nbase + t * 16 + nl;
      C[(size_t)row * N + col] = acc[t][r];
    }
}

// ---------------------------------------------------------------------------
// RoPE (rotate-half) + repack [S][nh*64] f32 -> [nh][S][64] f16
// ---------------------------------------------------------------------------
__global__ void rope_pack(const float* __restrict__ src,
                          const float* __restrict__ cosb,
                          const float* __restrict__ sinb,
                          _Float16* __restrict__ dst, int nh, int apply) {
  int i = blockIdx.x * blockDim.x + threadIdx.x;
  int total = S_LEN * nh * DHEAD;
  if (i >= total) return;
  int d = i & (DHEAD - 1);
  int h = (i / DHEAD) % nh;
  int s = i / (DHEAD * nh);
  int cols = nh * DHEAD;
  float v = src[(size_t)s * cols + h * DHEAD + d];
  if (apply) {
    float rot = (d < DHEAD / 2)
                  ? -src[(size_t)s * cols + h * DHEAD + d + DHEAD / 2]
                  :  src[(size_t)s * cols + h * DHEAD + d - DHEAD / 2];
    v = v * cosb[s * DHEAD + d] + rot * sinb[s * DHEAD + d];
  }
  dst[((size_t)h * S_LEN + s) * DHEAD + d] = (_Float16)v;
}

// ---------------------------------------------------------------------------
// Flash attention: grid (S/64, NQ); 128 threads = 4 waves.
// Each wave owns 16 query rows; online softmax over 64-key tiles.
// Output written as [s][H] f16 for the O-projection GEMM.
// ---------------------------------------------------------------------------
__global__ __launch_bounds__(128)
void flash_attn(const _Float16* __restrict__ q16,
                const _Float16* __restrict__ k16,
                const _Float16* __restrict__ v16,
                _Float16* __restrict__ o16) {
  __shared__ _Float16 Qs[64 * 64];
  __shared__ _Float16 Ks[64 * 64];
  __shared__ _Float16 Vs[64 * 64];
  __shared__ _Float16 Ps[64 * 64];

  const int qt    = blockIdx.x;           // query tile
  const int head  = blockIdx.y;
  const int g     = head / (NQ / NKV);    // kv head
  const int qbase = qt * 64;
  const int wave  = threadIdx.x >> 5;
  const int m0    = wave * 16;
  const int lane  = threadIdx.x & 31;
  const int hi    = lane >> 4;
  const int nl    = lane & 15;

  const _Float16* qg = q16 + (size_t)head * S_LEN * DHEAD;
  const _Float16* kg = k16 + (size_t)g    * S_LEN * DHEAD;
  const _Float16* vg = v16 + (size_t)g    * S_LEN * DHEAD;

  // Q tile: contiguous 8 KB, async into LDS (waited inside first iteration)
  async_tile64(qg + (size_t)qbase * DHEAD, DHEAD, Qs);

  float m_run[8], l_run[8];
  v8f oacc[4];
#pragma unroll
  for (int r = 0; r < 8; ++r) { m_run[r] = -1.0e30f; l_run[r] = 0.0f; }
#pragma unroll
  for (int t = 0; t < 4; ++t)
#pragma unroll
    for (int e = 0; e < 8; ++e) oacc[t][e] = 0.0f;

  for (int j = 0; j <= qt; ++j) {
    __syncthreads();                        // prior iteration reads done
    async_tile64(kg + (size_t)j * 64 * DHEAD, DHEAD, Ks);
    async_tile64(vg + (size_t)j * 64 * DHEAD, DHEAD, Vs);
    if (j < qt)  // hint next K tile toward GL2 (global_prefetch_b8)
      __builtin_prefetch(kg + (size_t)(j + 1) * 64 * DHEAD, 0, 1);
    wait_async_all();
    __syncthreads();

    // S = Q K^T  (16x64 per wave)
    v8f sacc[4];
#pragma unroll
    for (int t = 0; t < 4; ++t)
#pragma unroll
      for (int e = 0; e < 8; ++e) sacc[t][e] = 0.0f;
#pragma unroll
    for (int kc = 0; kc < 2; ++kc) {
      v16h a = wmma_load_a(Qs, 64, m0, kc * 32);
#pragma unroll
      for (int t = 0; t < 4; ++t) {
        v16h b = wmma_load_bt(Ks, 64, t * 16, kc * 32);
        sacc[t] = wmma_f16(a, b, sacc[t]);
      }
    }

    // scale + causal mask (only diagonal tile can violate causality)
#pragma unroll
    for (int t = 0; t < 4; ++t)
#pragma unroll
      for (int r = 0; r < 8; ++r) {
        float v = sacc[t][r] * SCALE;
        if (j == qt) {
          int key  = j * 64 + t * 16 + nl;
          int qrow = qbase + m0 + r + hi * 8;
          if (key > qrow) v = -1.0e30f;
        }
        sacc[t][r] = v;
      }

    // online softmax per row (rows live per-(r,half); reduce 16 lanes)
#pragma unroll
    for (int r = 0; r < 8; ++r) {
      float mx = sacc[0][r];
#pragma unroll
      for (int t = 1; t < 4; ++t) mx = fmaxf(mx, sacc[t][r]);
#pragma unroll
      for (int off = 1; off < 16; off <<= 1)
        mx = fmaxf(mx, __shfl_xor(mx, off, 32));
      float m_new = fmaxf(m_run[r], mx);
      float alpha = __expf(m_run[r] - m_new);
      float sum = 0.0f;
#pragma unroll
      for (int t = 0; t < 4; ++t) {
        float p = __expf(sacc[t][r] - m_new);
        sacc[t][r] = p;
        sum += p;
      }
#pragma unroll
      for (int off = 1; off < 16; off <<= 1)
        sum += __shfl_xor(sum, off, 32);
      l_run[r] = l_run[r] * alpha + sum;
      m_run[r] = m_new;
#pragma unroll
      for (int t = 0; t < 4; ++t) oacc[t][r] *= alpha;
    }

    // stage P (own rows only -> no cross-wave hazard)
#pragma unroll
    for (int t = 0; t < 4; ++t)
#pragma unroll
      for (int r = 0; r < 8; ++r)
        Ps[(m0 + r + hi * 8) * 64 + t * 16 + nl] = (_Float16)sacc[t][r];

    // O += P V
#pragma unroll
    for (int kc = 0; kc < 2; ++kc) {
      v16h a = wmma_load_a(Ps, 64, m0, kc * 32);
#pragma unroll
      for (int t = 0; t < 4; ++t) {
        v16h b = wmma_load_bn(Vs, 64, kc * 32, t * 16);
        oacc[t] = wmma_f16(a, b, oacc[t]);
      }
    }
  }

  // normalize + store to [s][H]
#pragma unroll
  for (int t = 0; t < 4; ++t)
#pragma unroll
    for (int r = 0; r < 8; ++r) {
      float inv = 1.0f / l_run[r];
      int row = qbase + m0 + r + hi * 8;
      int col = head * DHEAD + t * 16 + nl;
      o16[(size_t)row * HDIM + col] = (_Float16)(oacc[t][r] * inv);
    }
}

// ---------------------------------------------------------------------------
extern "C" void kernel_launch(void* const* d_in, const int* in_sizes, int n_in,
                              void* d_out, int out_size, void* d_ws, size_t ws_size,
                              hipStream_t stream) {
  const float* hs   = (const float*)d_in[0];
  const float* cosb = (const float*)d_in[1];
  const float* sinb = (const float*)d_in[2];
  const float* Wq   = (const float*)d_in[3];
  const float* Wk   = (const float*)d_in[4];
  const float* Wv   = (const float*)d_in[5];
  const float* Wo   = (const float*)d_in[6];
  float* out = (float*)d_out;

  char* ws = (char*)d_ws;
  size_t off = 0;
  auto carve = [&](size_t bytes) {
    void* p = ws + off;
    off += (bytes + 255) & ~(size_t)255;
    return p;
  };
  _Float16* hs16 = (_Float16*)carve((size_t)S_LEN * HDIM * 2);
  _Float16* Wq16 = (_Float16*)carve((size_t)HDIM * HDIM * 2);
  _Float16* Wk16 = (_Float16*)carve((size_t)KVDIM * HDIM * 2);
  _Float16* Wv16 = (_Float16*)carve((size_t)KVDIM * HDIM * 2);
  _Float16* Wo16 = (_Float16*)carve((size_t)HDIM * HDIM * 2);
  float*    qf   = (float*)carve((size_t)S_LEN * HDIM * 4);
  float*    kf   = (float*)carve((size_t)S_LEN * KVDIM * 4);
  float*    vf   = (float*)carve((size_t)S_LEN * KVDIM * 4);
  _Float16* q16  = (_Float16*)carve((size_t)S_LEN * HDIM * 2);
  _Float16* k16  = (_Float16*)carve((size_t)S_LEN * KVDIM * 2);
  _Float16* v16  = (_Float16*)carve((size_t)S_LEN * KVDIM * 2);
  _Float16* a16  = (_Float16*)carve((size_t)S_LEN * HDIM * 2);

  auto cvt = [&](const float* s, _Float16* d, int n) {
    cvt_f32_f16<<<(n + 255) / 256, 256, 0, stream>>>(s, d, n);
  };
  cvt(hs, hs16, S_LEN * HDIM);
  cvt(Wq, Wq16, HDIM * HDIM);
  cvt(Wk, Wk16, KVDIM * HDIM);
  cvt(Wv, Wv16, KVDIM * HDIM);
  cvt(Wo, Wo16, HDIM * HDIM);

  // projections: X @ W^T
  gemm_nt_f16<<<dim3(HDIM / 64, S_LEN / 64), 128, 0, stream>>>(
      hs16, Wq16, qf, S_LEN, HDIM, HDIM);
  gemm_nt_f16<<<dim3(KVDIM / 64, S_LEN / 64), 128, 0, stream>>>(
      hs16, Wk16, kf, S_LEN, KVDIM, HDIM);
  gemm_nt_f16<<<dim3(KVDIM / 64, S_LEN / 64), 128, 0, stream>>>(
      hs16, Wv16, vf, S_LEN, KVDIM, HDIM);

  // RoPE + repack to [head][s][d]
  rope_pack<<<(S_LEN * NQ * DHEAD + 255) / 256, 256, 0, stream>>>(
      qf, cosb, sinb, q16, NQ, 1);
  rope_pack<<<(S_LEN * NKV * DHEAD + 255) / 256, 256, 0, stream>>>(
      kf, cosb, sinb, k16, NKV, 1);
  rope_pack<<<(S_LEN * NKV * DHEAD + 255) / 256, 256, 0, stream>>>(
      vf, cosb, sinb, v16, NKV, 0);

  // causal GQA flash attention
  flash_attn<<<dim3(S_LEN / 64, NQ), 128, 0, stream>>>(q16, k16, v16, a16);

  // output projection
  gemm_nt_f16<<<dim3(HDIM / 64, S_LEN / 64), 128, 0, stream>>>(
      a16, Wo16, out, S_LEN, HDIM, HDIM);
}
